// SwinTransformerLayer_19189913879178
// MI455X (gfx1250) — compile-verified, matrix operand order checked by reading, not run on hardware
//
#include <hip/hip_runtime.h>

#define IMG   128
#define CDIM  192
#define NHEAD 6
#define HD    32
#define WSZ   8
#define SHFT  4
#define HIDD  768

typedef __attribute__((ext_vector_type(16))) _Float16     v16h;
typedef __attribute__((ext_vector_type(8)))  float        v8f;
typedef __attribute__((ext_vector_type(8)))  unsigned int v8u;
typedef __attribute__((ext_vector_type(4)))  int          v4i;

typedef __attribute__((address_space(1))) v4i* global_v4i_p;
typedef __attribute__((address_space(3))) v4i* shared_v4i_p;

// LDS leading dimensions (halves; all even, 16B-aligned rows where staged)
#define LDW  200   // ln'd activations / attention output (64 x 192)
#define LDQ  392   // q,k                                  (64 x 384)
#define LDV  72    // V transposed  [dim][token]           (192 x 64)
#define LDP  72    // softmax probabilities                (64 x 64)
#define LDT  48    // staged transposed B tile             (64 x 32) rows=96B
#define LDSC 68    // f32 score matrix                     (64 x 64)
#define LDH  200   // ln2'd activations                    (64 x 192)
#define LDG  72    // gelu'd hidden chunk                  (64 x 64)
#define LDB2 72    // staged w2T chunk [out][k]            (192 x 64) rows=144B

#if defined(__has_builtin)
#if __has_builtin(__builtin_amdgcn_global_load_async_to_lds_b128)
#define ASYNC_LDS 1
#endif
#endif

__device__ __forceinline__ void async_copy16(const _Float16* g, _Float16* l) {
#ifdef ASYNC_LDS
  __builtin_amdgcn_global_load_async_to_lds_b128((global_v4i_p)g, (shared_v4i_p)l, 0,
                                                 0);
#else
  *(uint4*)l = *(const uint4*)g;
#endif
}
__device__ __forceinline__ void async_wait() {
#ifdef ASYNC_LDS
#if __has_builtin(__builtin_amdgcn_s_wait_asynccnt)
  __builtin_amdgcn_s_wait_asynccnt(0);
#else
  asm volatile("s_wait_asynccnt 0" ::: "memory");
#endif
#endif
}

__device__ __forceinline__ v8f wmma16(v16h a, v16h b, v8f c) {
  return __builtin_amdgcn_wmma_f32_16x16x32_f16(false, a, false, b, (short)0, c,
                                                false, false);
}

// 16x32 f16 operand fragment from row-major [rows][ld] at (row0, k0).
// Lane l: row = l&15, K-half = (l>>4)*8; VGPR q holds K pair -> one b32 load.
// Serves as A (row=M) and, for N x K sources, as B (row=N).
__device__ __forceinline__ v16h frag_a(const _Float16* base, int row0, int k0, int ld) {
  int lane = threadIdx.x & 31;
  int m = lane & 15;
  int kh = (lane >> 4) << 3;
  const unsigned int* p = (const unsigned int*)(base + (row0 + m) * ld + k0 + kh);
  v8u u;
#pragma unroll
  for (int q = 0; q < 8; ++q) {
    int kb2 = (q & 3) + ((q >> 2) << 3);  // pair index
    u[q] = p[kb2];
  }
  return __builtin_bit_cast(v16h, u);
}

// C/D (16x16 f32): VGPR r, lane l -> (M = r + 8*(l>>4), N = l&15)
__device__ __forceinline__ void store_c_f32(float* base, int row0, int col0, int ld,
                                            v8f c) {
  int lane = threadIdx.x & 31;
  int n = lane & 15;
  int mh = (lane >> 4) << 3;
#pragma unroll
  for (int r = 0; r < 8; ++r) base[(row0 + mh + r) * ld + col0 + n] = c[r];
}

__device__ __forceinline__ int region9(int gh, int gw) {
  int rh = (gh < IMG - WSZ) ? 0 : ((gh < IMG - SHFT) ? 1 : 2);
  int rw = (gw < IMG - WSZ) ? 0 : ((gw < IMG - SHFT) ? 1 : 2);
  return rh * 3 + rw;
}

// Stage a 64(rows) x 32(k) tile of a transposed (N x K) f16 weight into LDS.
__device__ __forceinline__ void stage_tile(const _Float16* __restrict__ wT, int row0,
                                           int gld, int k0, _Float16* sBt) {
  int idx = threadIdx.x * 8;  // halves
  int row = idx >> 5;         // 0..63
  int col = idx & 31;
  const _Float16* g = wT + (size_t)(row0 + row) * gld + k0 + col;
  async_copy16(g, sBt + row * LDT + col);
  __builtin_prefetch(g + 32, 0, 1);  // next K tile -> global_prefetch_b8
}

// --------------------------------------------------------------------------
// f32 -> f16 transposed weight conversion: a is K x N, b becomes N x K.
// --------------------------------------------------------------------------
__global__ void cvt_f16_T_kernel(const float* __restrict__ a, _Float16* __restrict__ b,
                                 int K, int N) {
  int i = blockIdx.x * 256 + threadIdx.x;
  if (i < K * N) {
    int k = i / N, n = i - k * N;
    b[(size_t)n * K + k] = (_Float16)a[i];
  }
}

// --------------------------------------------------------------------------
// Kernel A: LN1 + shifted-window attention + proj + residual  (block = window)
// --------------------------------------------------------------------------
__global__ __launch_bounds__(256) void swin_attn_kernel(
    const float* __restrict__ x, const float* __restrict__ n1w,
    const float* __restrict__ n1b, const _Float16* __restrict__ qkv_wT,
    const float* __restrict__ qkv_b, const float* __restrict__ rel_bias,
    const _Float16* __restrict__ proj_wT, const float* __restrict__ proj_b,
    float* __restrict__ x1) {
  extern __shared__ char smem[];
  float*    sScore = (float*)smem;                    // 64 x LDSC f32
  _Float16* sW     = (_Float16*)(sScore + 64 * LDSC); // 64 x LDW
  _Float16* sQKV   = sW + 64 * LDW;                   // 64 x LDQ (q,k)
  _Float16* sVt    = sQKV + 64 * LDQ;                 // 192 x LDV (v^T)
  _Float16* sP     = sVt + 192 * LDV;                 // 64 x LDP
  _Float16* sBt    = sP + 64 * LDP;                   // 64 x LDT staging
  _Float16* sOut   = sW;                              // aliases sW (dead then)

  const int wid  = blockIdx.x;
  const int b    = wid >> 8;
  const int win  = wid & 255;
  const int wr   = win >> 4;
  const int wc   = win & 15;
  const int bimg = b * IMG * IMG;
  const int tid  = threadIdx.x;
  const int wave = tid >> 5;
  const int lane = tid & 31;

  // ---- Stage 1: gather shifted window + LayerNorm1 -> sW (f16) ----
  {
    int tok  = tid >> 2;
    int part = tid & 3;
    int tr = tok >> 3, tc = tok & 7;
    int oh = (wr * WSZ + tr + SHFT) & (IMG - 1);
    int ow = (wc * WSZ + tc + SHFT) & (IMG - 1);
    const float* xp = x + ((size_t)(bimg + oh * IMG + ow)) * CDIM + part * 48;
    float v[48];
    float s = 0.f, s2 = 0.f;
#pragma unroll
    for (int i = 0; i < 48; ++i) {
      float t = xp[i];
      v[i] = t;
      s += t;
      s2 += t * t;
    }
    s  += __shfl_xor(s, 1);  s  += __shfl_xor(s, 2);
    s2 += __shfl_xor(s2, 1); s2 += __shfl_xor(s2, 2);
    float mean = s * (1.f / CDIM);
    float var  = s2 * (1.f / CDIM) - mean * mean;
    float rstd = rsqrtf(var + 1e-5f);
    _Float16* wp = sW + tok * LDW + part * 48;
#pragma unroll
    for (int i = 0; i < 48; ++i) {
      int ch = part * 48 + i;
      wp[i] = (_Float16)((v[i] - mean) * rstd * n1w[ch] + n1b[ch]);
    }
  }
  __syncthreads();

  // ---- Stage 2: QKV GEMM  (64x192) @ (192x576); weights are N x K ----
  {
    int t0 = wave * 2, t1 = t0 + 1;  // mt = t>>2, nt = t&3 per 64-col chunk
    int n = lane & 15, mh = (lane >> 4) << 3;
    for (int nc = 0; nc < 9; ++nc) {
      int n0 = nc * 64;
      v8f acc0 = {}, acc1 = {};
      for (int ks = 0; ks < 6; ++ks) {
        int k0 = ks * 32;
        __syncthreads();
        stage_tile(qkv_wT, n0, CDIM, k0, sBt);
        async_wait();
        __syncthreads();
        v16h a0 = frag_a(sW, (t0 >> 2) * 16, k0, LDW);
        v16h b0 = frag_a(sBt, (t0 & 3) * 16, 0, LDT);
        acc0 = wmma16(a0, b0, acc0);
        v16h a1 = frag_a(sW, (t1 >> 2) * 16, k0, LDW);
        v16h b1 = frag_a(sBt, (t1 & 3) * 16, 0, LDT);
        acc1 = wmma16(a1, b1, acc1);
      }
      auto qkv_epi = [&](v8f acc, int t) {
        int col = n0 + (t & 3) * 16 + n;
        float bias = qkv_b[col];
        if (col < 2 * CDIM) {  // q (scaled) and k, row-major [tok][col]
          float scl = (col < CDIM) ? 0.17677669529663687f : 1.0f;  // hd^-0.5
#pragma unroll
          for (int r = 0; r < 8; ++r) {
            int row = (t >> 2) * 16 + mh + r;
            sQKV[row * LDQ + col] = (_Float16)((acc[r] + bias) * scl);
          }
        } else {  // v transposed: [dim][tok] (contiguous over r)
          int d = col - 2 * CDIM;
#pragma unroll
          for (int r = 0; r < 8; ++r) {
            int row = (t >> 2) * 16 + mh + r;
            sVt[d * LDV + row] = (_Float16)(acc[r] + bias);
          }
        }
      };
      qkv_epi(acc0, t0);
      qkv_epi(acc1, t1);
    }
  }
  __syncthreads();

  // ---- Stage 3: per-head attention ----
  for (int h = 0; h < NHEAD; ++h) {
    {  // scores = q @ k^T (64x64, K=32): 16 tiles, 2 per wave
      int t0 = wave * 2, t1 = t0 + 1;
      v16h aq0 = frag_a(sQKV, (t0 >> 2) * 16, h * HD, LDQ);
      v16h bk0 = frag_a(sQKV, (t0 & 3) * 16, CDIM + h * HD, LDQ);
      v8f c0 = {};
      c0 = wmma16(aq0, bk0, c0);
      v16h aq1 = frag_a(sQKV, (t1 >> 2) * 16, h * HD, LDQ);
      v16h bk1 = frag_a(sQKV, (t1 & 3) * 16, CDIM + h * HD, LDQ);
      v8f c1 = {};
      c1 = wmma16(aq1, bk1, c1);
      store_c_f32(sScore, (t0 >> 2) * 16, (t0 & 3) * 16, LDSC, c0);
      store_c_f32(sScore, (t1 >> 2) * 16, (t1 & 3) * 16, LDSC, c1);
    }
    __syncthreads();
    {  // softmax + rel-pos bias + shift mask (4 threads per row)
      int row  = tid >> 2;
      int part = tid & 3;
      int m0 = part * 16;
      int qr = row >> 3, qc = row & 7;
      int qreg = region9(wr * WSZ + qr, wc * WSZ + qc);
      float rv[16];
      float mx = -3.0e38f;
#pragma unroll
      for (int i = 0; i < 16; ++i) {
        int m = m0 + i;
        int kr = m >> 3, kc = m & 7;
        int rel = (qr - kr + WSZ - 1) * (2 * WSZ - 1) + (qc - kc + WSZ - 1);
        float bias = rel_bias[rel * NHEAD + h];
        int kreg = region9(wr * WSZ + kr, wc * WSZ + kc);
        float msk = (qreg == kreg) ? 0.f : -100.f;
        float vv = sScore[row * LDSC + m] + bias + msk;
        rv[i] = vv;
        mx = fmaxf(mx, vv);
      }
      mx = fmaxf(mx, __shfl_xor(mx, 1));
      mx = fmaxf(mx, __shfl_xor(mx, 2));
      float sum = 0.f;
#pragma unroll
      for (int i = 0; i < 16; ++i) {
        rv[i] = __expf(rv[i] - mx);
        sum += rv[i];
      }
      sum += __shfl_xor(sum, 1);
      sum += __shfl_xor(sum, 2);
      float inv = 1.f / sum;
#pragma unroll
      for (int i = 0; i < 16; ++i)
        sP[row * LDP + m0 + i] = (_Float16)(rv[i] * inv);
    }
    __syncthreads();
    {  // out_head = P(64x64) @ V(64x32): 8 tiles, 1 per wave; B from sVt (N x K)
      int mt = wave >> 1, nt = wave & 1;
      v8f c = {};
#pragma unroll
      for (int ks = 0; ks < 2; ++ks) {
        v16h a  = frag_a(sP, mt * 16, ks * 32, LDP);
        v16h bv = frag_a(sVt, h * HD + nt * 16, ks * 32, LDV);
        c = wmma16(a, bv, c);
      }
      int n = lane & 15, mh = (lane >> 4) << 3;
#pragma unroll
      for (int r = 0; r < 8; ++r)
        sOut[(mt * 16 + mh + r) * LDW + h * HD + nt * 16 + n] = (_Float16)c[r];
    }
    __syncthreads();
  }

  // ---- Stage 4: proj GEMM + residual, scatter to un-shifted coords ----
  {
    int t0 = wave * 2, t1 = t0 + 1;
    int n = lane & 15, mh = (lane >> 4) << 3;
    for (int nc = 0; nc < 3; ++nc) {
      int n0 = nc * 64;
      v8f acc0 = {}, acc1 = {};
      for (int ks = 0; ks < 6; ++ks) {
        int k0 = ks * 32;
        __syncthreads();
        stage_tile(proj_wT, n0, CDIM, k0, sBt);
        async_wait();
        __syncthreads();
        v16h a0 = frag_a(sOut, (t0 >> 2) * 16, k0, LDW);
        v16h b0 = frag_a(sBt, (t0 & 3) * 16, 0, LDT);
        acc0 = wmma16(a0, b0, acc0);
        v16h a1 = frag_a(sOut, (t1 >> 2) * 16, k0, LDW);
        v16h b1 = frag_a(sBt, (t1 & 3) * 16, 0, LDT);
        acc1 = wmma16(a1, b1, acc1);
      }
      auto proj_epi = [&](v8f acc, int t) {
        int col = n0 + (t & 3) * 16 + n;
        float pb = proj_b[col];
#pragma unroll
        for (int r = 0; r < 8; ++r) {
          int tok = (t >> 2) * 16 + mh + r;
          int tr = tok >> 3, tc = tok & 7;
          int oh = (wr * WSZ + tr + SHFT) & (IMG - 1);
          int ow = (wc * WSZ + tc + SHFT) & (IMG - 1);
          size_t gi = ((size_t)(bimg + oh * IMG + ow)) * CDIM + col;
          x1[gi] = x[gi] + acc[r] + pb;
        }
      };
      proj_epi(acc0, t0);
      proj_epi(acc1, t1);
    }
  }
}

// --------------------------------------------------------------------------
// Kernel B: LN2 + MLP(192->768->192, exact GELU) + residual (block = 64 rows)
// --------------------------------------------------------------------------
__global__ __launch_bounds__(256) void swin_mlp_kernel(
    const float* __restrict__ x1, const float* __restrict__ n2w,
    const float* __restrict__ n2b, const _Float16* __restrict__ w1T,
    const float* __restrict__ b1, const _Float16* __restrict__ w2T,
    const float* __restrict__ b2, float* __restrict__ out) {
  extern __shared__ char smem[];
  _Float16* sH  = (_Float16*)smem;       // 64 x LDH
  _Float16* sG  = sH + 64 * LDH;         // 64 x LDG
  _Float16* sBt = sG + 64 * LDG;         // 64 x LDT staging (w1T tile)
  _Float16* sB2 = sBt + 64 * LDT;        // 192 x LDB2 (w2T chunk)

  const int tid  = threadIdx.x;
  const int wave = tid >> 5;
  const int lane = tid & 31;
  const size_t row0 = (size_t)blockIdx.x * 64;

  // ---- LN2 -> sH (f16) ----
  {
    int tok = tid >> 2, part = tid & 3;
    const float* xp = x1 + (row0 + tok) * CDIM + part * 48;
    float v[48];
    float s = 0.f, s2 = 0.f;
#pragma unroll
    for (int i = 0; i < 48; ++i) {
      float t = xp[i];
      v[i] = t;
      s += t;
      s2 += t * t;
    }
    s  += __shfl_xor(s, 1);  s  += __shfl_xor(s, 2);
    s2 += __shfl_xor(s2, 1); s2 += __shfl_xor(s2, 2);
    float mean = s * (1.f / CDIM);
    float var  = s2 * (1.f / CDIM) - mean * mean;
    float rstd = rsqrtf(var + 1e-5f);
    _Float16* hp = sH + tok * LDH + part * 48;
#pragma unroll
    for (int i = 0; i < 48; ++i) {
      int ch = part * 48 + i;
      hp[i] = (_Float16)((v[i] - mean) * rstd * n2w[ch] + n2b[ch]);
    }
  }
  __syncthreads();

  v8f oacc[6];  // 48 out tiles (mt 0..3, nt 0..11), wave owns 6
#pragma unroll
  for (int i = 0; i < 6; ++i) oacc[i] = (v8f){};

  for (int hc = 0; hc < 12; ++hc) {
    // hidden tile = sH @ w1[:, hc*64..+64]; w1T is [hidden][cdim]
    int t0 = wave * 2, t1 = t0 + 1;
    v8f h0 = {}, h1 = {};
    for (int ks = 0; ks < 6; ++ks) {
      int k0 = ks * 32;
      __syncthreads();
      stage_tile(w1T, hc * 64, CDIM, k0, sBt);
      async_wait();
      __syncthreads();
      v16h a0 = frag_a(sH, (t0 >> 2) * 16, k0, LDH);
      v16h b0 = frag_a(sBt, (t0 & 3) * 16, 0, LDT);
      h0 = wmma16(a0, b0, h0);
      v16h a1 = frag_a(sH, (t1 >> 2) * 16, k0, LDH);
      v16h b1 = frag_a(sBt, (t1 & 3) * 16, 0, LDT);
      h1 = wmma16(a1, b1, h1);
    }
    __syncthreads();  // prior reads of sG / sB2 finished
    {  // bias + exact GELU -> sG (f16)
      int n = lane & 15, mh = (lane >> 4) << 3;
      auto gelu_epi = [&](v8f acc, int t) {
        int col = (t & 3) * 16 + n;
        float bb = b1[hc * 64 + col];
#pragma unroll
        for (int r = 0; r < 8; ++r) {
          float v = acc[r] + bb;
          float g = 0.5f * v * (1.f + erff(v * 0.70710678118654752f));
          sG[((t >> 2) * 16 + mh + r) * LDG + col] = (_Float16)g;
        }
      };
      gelu_epi(h0, t0);
      gelu_epi(h1, t1);
    }
    {  // stage w2T chunk: rows = out-col n (192), k = hidden in chunk (64)
#pragma unroll
      for (int i = 0; i < 6; ++i) {
        int cid = tid + 256 * i;  // 0..1535 16B-chunks
        int row = cid >> 3;       // 0..191
        int seg = cid & 7;
        async_copy16(w2T + (size_t)row * HIDD + hc * 64 + seg * 8,
                     sB2 + row * LDB2 + seg * 8);
      }
      async_wait();
    }
    __syncthreads();
    // oacc += sG(64x64) @ w2_chunk; B from sB2 (N x K)
#pragma unroll
    for (int i = 0; i < 6; ++i) {
      int t = wave * 6 + i;
      int mt = t / 12, nt = t % 12;
#pragma unroll
      for (int ks = 0; ks < 2; ++ks) {
        v16h a = frag_a(sG, mt * 16, ks * 32, LDG);
        v16h b = frag_a(sB2, nt * 16, ks * 32, LDB2);
        oacc[i] = wmma16(a, b, oacc[i]);
      }
    }
  }
  __syncthreads();

  // ---- final: out = x1 + mlp + b2 ----
  {
    int n = lane & 15, mh = (lane >> 4) << 3;
#pragma unroll
    for (int i = 0; i < 6; ++i) {
      int t = wave * 6 + i;
      int mt = t / 12, nt = t % 12;
      int col = nt * 16 + n;
      float bb = b2[col];
#pragma unroll
      for (int r = 0; r < 8; ++r) {
        size_t gi = (row0 + mt * 16 + mh + r) * CDIM + col;
        out[gi] = x1[gi] + oacc[i][r] + bb;
      }
    }
  }
}

// --------------------------------------------------------------------------
extern "C" void kernel_launch(void* const* d_in, const int* in_sizes, int n_in,
                              void* d_out, int out_size, void* d_ws, size_t ws_size,
                              hipStream_t stream) {
  const float* x      = (const float*)d_in[0];
  const float* n1w    = (const float*)d_in[1];
  const float* n1b    = (const float*)d_in[2];
  const float* qkv_w  = (const float*)d_in[3];
  const float* qkv_b  = (const float*)d_in[4];
  const float* relb   = (const float*)d_in[5];
  const float* proj_w = (const float*)d_in[6];
  const float* proj_b = (const float*)d_in[7];
  const float* n2w    = (const float*)d_in[8];
  const float* n2b    = (const float*)d_in[9];
  const float* w1     = (const float*)d_in[10];
  const float* b1     = (const float*)d_in[11];
  const float* w2     = (const float*)d_in[12];
  const float* b2     = (const float*)d_in[13];

  char* ws = (char*)d_ws;
  _Float16* qkv_wT  = (_Float16*)ws;                 // 576 x 192
  _Float16* proj_wT = qkv_wT + CDIM * 3 * CDIM;      // 192 x 192
  _Float16* w1T     = proj_wT + CDIM * CDIM;         // 768 x 192
  _Float16* w2T     = w1T + CDIM * HIDD;             // 192 x 768
  float*    x1      = (float*)(ws + (1 << 20));      // 131072 x 192 f32

  int nq = CDIM * 3 * CDIM, np = CDIM * CDIM, n1 = CDIM * HIDD, n2 = HIDD * CDIM;
  cvt_f16_T_kernel<<<(nq + 255) / 256, 256, 0, stream>>>(qkv_w, qkv_wT, CDIM,
                                                         3 * CDIM);
  cvt_f16_T_kernel<<<(np + 255) / 256, 256, 0, stream>>>(proj_w, proj_wT, CDIM, CDIM);
  cvt_f16_T_kernel<<<(n1 + 255) / 256, 256, 0, stream>>>(w1, w1T, CDIM, HIDD);
  cvt_f16_T_kernel<<<(n2 + 255) / 256, 256, 0, stream>>>(w2, w2T, HIDD, CDIM);

  size_t smemA = (size_t)64 * LDSC * 4 +
                 (size_t)2 * (64 * LDW + 64 * LDQ + 192 * LDV + 64 * LDP + 64 * LDT);
  size_t smemB = (size_t)2 * (64 * LDH + 64 * LDG + 64 * LDT + 192 * LDB2);

  swin_attn_kernel<<<2048, 256, smemA, stream>>>(x, n1w, n1b, qkv_wT, qkv_b, relb,
                                                 proj_wT, proj_b, x1);
  swin_mlp_kernel<<<2048, 256, smemB, stream>>>(x1, n2w, n2b, w1T, b1, w2T, b2,
                                                (float*)d_out);
}
